// DiT_9259949490457
// MI455X (gfx1250) — matrix operand order repaired and checked by reading, not compiled, not run on hardware
//
#include <hip/hip_runtime.h>
#include <stdint.h>

// ---------------------------------------------------------------------------
// Focused linear attention DiT block on gfx1250 (MI455X).
// bf16 WMMA (v_wmma_f32_16x16x32_bf16) for all GEMMs, async global->LDS
// copies (global_load_async_to_lds_b128 + s_wait_asynccnt) with double
// buffering, wave32 fragments, A/B in contiguous k order (k = 16*(lane/16) + s).
// ---------------------------------------------------------------------------

typedef __bf16 bf16_t;
typedef __bf16 v16bf __attribute__((ext_vector_type(16)));
typedef __bf16 v8bf  __attribute__((ext_vector_type(8)));
typedef float  v8f   __attribute__((ext_vector_type(8)));

#define DIM   1152
#define NHEAD 12
#define KVH   4
#define HD    96
#define BATCH 16
#define NTOK  1024
#define MROWS (BATCH * NTOK)   // 16384

__device__ __forceinline__ v8f wmma_bf16(v16bf a, v16bf b, v8f c) {
  // D = A(16x32) * B(32x16) + C, f32 accum
  return __builtin_amdgcn_wmma_f32_16x16x32_bf16(
      /*neg_a=*/false, a, /*neg_b=*/false, b,
      /*c_mod=*/(short)0, c, /*reuse_a=*/false, /*reuse_b=*/false);
}

__device__ __forceinline__ uint32_t lds_addr(const void* p) {
  // low 32 bits of a flat LDS-aperture address are the LDS byte offset
  return (uint32_t)(uintptr_t)p;
}

__device__ __forceinline__ void async_b128(uint32_t lds, const bf16_t* g) {
  asm volatile("global_load_async_to_lds_b128 %0, %1, off"
               :: "v"(lds), "v"((unsigned long long)(uintptr_t)g)
               : "memory");
}

__device__ __forceinline__ void wait_async0() {
  asm volatile("s_wait_asynccnt 0x0" ::: "memory");
}

// Build a 16-half fragment from two 16B LDS chunks (elements 0..7 / 8..15).
__device__ __forceinline__ v16bf frag16(const bf16_t* p0, const bf16_t* p1) {
  v8bf lo = *(const v8bf*)p0;
  v8bf hi = *(const v8bf*)p1;
  v16bf r;
#pragma unroll
  for (int i = 0; i < 8; ++i) { r[i] = lo[i]; r[i + 8] = hi[i]; }
  return r;
}

// ---------------------------------------------------------------------------
// f32 -> bf16 convert
// ---------------------------------------------------------------------------
__global__ __launch_bounds__(256) void cvt_bf16(const float* __restrict__ src,
                                                bf16_t* __restrict__ dst, int n) {
  int i = blockIdx.x * 256 + threadIdx.x;
  if (i < n) dst[i] = (bf16_t)src[i];
}

// ---------------------------------------------------------------------------
// Main GEMM: C[M,N] = A[M,K](bf16) * W[N,K]^T(bf16) + bias, f32 out.
// Block tile 128x64, BK=32, 8 waves (4 M x 2 N), 2x2 WMMA tiles per wave.
// Async global->LDS double buffering.
// ---------------------------------------------------------------------------
__global__ __launch_bounds__(256) void gemm_nt_bias_f32(
    const bf16_t* __restrict__ A, const bf16_t* __restrict__ W,
    const float* __restrict__ bias, float* __restrict__ C,
    int M, int N, int K) {
  __shared__ __align__(16) bf16_t sA[2][128][32];
  __shared__ __align__(16) bf16_t sB[2][64][32];

  const int nblk = N / 64;
  const int bm = (int)blockIdx.x / nblk;
  const int bn = (int)blockIdx.x % nblk;
  const int row0 = bm * 128, col0 = bn * 64;

  const int t = threadIdx.x;
  const int lane = t & 31;
  const int wid = t >> 5;
  const int wm = wid & 3;        // 0..3 -> 32-row strip
  const int wn = wid >> 2;       // 0..1 -> 32-col strip

  auto stage = [&](int k0, int buf) {
    // A: 512 x 16B chunks, 2 per thread
#pragma unroll
    for (int i = 0; i < 2; ++i) {
      int c = t + i * 256;
      int r = c >> 2, c16 = c & 3;
      async_b128(lds_addr(&sA[buf][r][c16 * 8]),
                 A + (size_t)(row0 + r) * K + k0 + c16 * 8);
    }
    // W: 256 x 16B chunks, 1 per thread
    {
      int r = t >> 2, c16 = t & 3;
      async_b128(lds_addr(&sB[buf][r][c16 * 8]),
                 W + (size_t)(col0 + r) * K + k0 + c16 * 8);
    }
  };

  v8f acc[2][2] = {};
  stage(0, 0);
  wait_async0();
  __syncthreads();

  const int KT = K / 32;
  const int mrow = lane & 15;
  const int hi = lane >> 4;
  const int abase = hi * 16;   // A: K-halves {abase..abase+15}
  const int kb = hi * 16;      // B: K-halves {kb..kb+15}

  for (int kt = 0; kt < KT; ++kt) {
    int buf = kt & 1;
    if (kt + 1 < KT) stage((kt + 1) * 32, buf ^ 1);

    v16bf af[2], bf[2];
#pragma unroll
    for (int i = 0; i < 2; ++i)
      af[i] = frag16(&sA[buf][wm * 32 + i * 16 + mrow][abase],
                     &sA[buf][wm * 32 + i * 16 + mrow][abase + 8]);
#pragma unroll
    for (int j = 0; j < 2; ++j)
      bf[j] = frag16(&sB[buf][wn * 32 + j * 16 + mrow][kb],
                     &sB[buf][wn * 32 + j * 16 + mrow][kb + 8]);
#pragma unroll
    for (int i = 0; i < 2; ++i)
#pragma unroll
      for (int j = 0; j < 2; ++j)
        acc[i][j] = wmma_bf16(af[i], bf[j], acc[i][j]);

    wait_async0();
    __syncthreads();
  }

  // C/D layout: col = lane&15, row = r + 8*(lane>>4)
#pragma unroll
  for (int i = 0; i < 2; ++i)
#pragma unroll
    for (int j = 0; j < 2; ++j) {
      int col = col0 + wn * 32 + j * 16 + mrow;
      float bv = bias ? bias[col] : 0.f;
#pragma unroll
      for (int r = 0; r < 8; ++r) {
        int rowg = row0 + wm * 32 + i * 16 + r + hi * 8;
        C[(size_t)rowg * N + col] = acc[i][j][r] + bv;
      }
    }
}

// ---------------------------------------------------------------------------
// focus(q): per (b,h) block: scale = ||relu(a)||_F / ||relu(a)^3||_F,
// write bf16 scale*relu(a)^3, layout [b,h][N][HD].
// ---------------------------------------------------------------------------
__global__ __launch_bounds__(256) void focus_q(const float* __restrict__ Qpre,
                                               bf16_t* __restrict__ qf) {
  const int bh = blockIdx.x;             // 0..191
  const int b = bh / NHEAD, h = bh % NHEAD;
  const float* src = Qpre + (size_t)b * NTOK * DIM + h * HD;
  const int t = threadIdx.x;

  float s2 = 0.f, s6 = 0.f;
  for (int i = t; i < NTOK * HD; i += 256) {
    int n = i / HD, d = i - n * HD;
    float a = fmaxf(src[(size_t)n * DIM + d], 0.f);
    float a2 = a * a, a3 = a2 * a;
    s2 += a2;
    s6 += a3 * a3;
  }
  __shared__ float red[2][256];
  red[0][t] = s2; red[1][t] = s6;
  __syncthreads();
  for (int s = 128; s > 0; s >>= 1) {
    if (t < s) { red[0][t] += red[0][t + s]; red[1][t] += red[1][t + s]; }
    __syncthreads();
  }
  float scale = sqrtf(red[0][0]) / fmaxf(sqrtf(red[1][0]), 1e-12f);

  bf16_t* dst = qf + (size_t)bh * NTOK * HD;
  for (int i = t; i < NTOK * HD; i += 256) {
    int n = i / HD, d = i - n * HD;
    float a = fmaxf(src[(size_t)n * DIM + d], 0.f);
    dst[i] = (bf16_t)(scale * a * a * a);
  }
}

// ---------------------------------------------------------------------------
// focus(k) transposed + v transpose: per (b,kvh) block.
// kfT/vT layout: [b,kvh][HD][N] (K-contiguous in token dim for WMMA).
// ---------------------------------------------------------------------------
__global__ __launch_bounds__(256) void focus_kv(const float* __restrict__ KVpre,
                                                bf16_t* __restrict__ kfT,
                                                bf16_t* __restrict__ vT) {
  const int bh = blockIdx.x;             // 0..63
  const int b = bh / KVH, g = bh % KVH;
  const float* ksrc = KVpre + (size_t)b * NTOK * (2 * KVH * HD) + g * HD;
  const float* vsrc = ksrc + KVH * HD;   // v block starts at col 384
  const int t = threadIdx.x;
  const int kvdim = 2 * KVH * HD;        // 768 row stride

  float s2 = 0.f, s6 = 0.f;
  for (int i = t; i < NTOK * HD; i += 256) {
    int n = i / HD, d = i - n * HD;
    float a = fmaxf(ksrc[(size_t)n * kvdim + d], 0.f);
    float a2 = a * a, a3 = a2 * a;
    s2 += a2;
    s6 += a3 * a3;
  }
  __shared__ float red[2][256];
  red[0][t] = s2; red[1][t] = s6;
  __syncthreads();
  for (int s = 128; s > 0; s >>= 1) {
    if (t < s) { red[0][t] += red[0][t + s]; red[1][t] += red[1][t + s]; }
    __syncthreads();
  }
  float scale = sqrtf(red[0][0]) / fmaxf(sqrtf(red[1][0]), 1e-12f);

  bf16_t* kdst = kfT + (size_t)bh * HD * NTOK;
  bf16_t* vdst = vT + (size_t)bh * HD * NTOK;
  for (int i = t; i < NTOK * HD; i += 256) {
    int n = i / HD, d = i - n * HD;
    float a = fmaxf(ksrc[(size_t)n * kvdim + d], 0.f);
    kdst[(size_t)d * NTOK + n] = (bf16_t)(scale * a * a * a);
    vdst[(size_t)d * NTOK + n] = (bf16_t)vsrc[(size_t)n * kvdim + d];
  }
}

// ---------------------------------------------------------------------------
// kvT[e][d] = sum_n v[n,e] * k_f[n,d]   (per (b,kvh); M=N=96, K=1024)
// 288 threads = 9 waves in a 3x3 grid, 2x2 WMMA tiles each.
// ---------------------------------------------------------------------------
__global__ __launch_bounds__(288) void kv_gemm(const bf16_t* __restrict__ vT,
                                               const bf16_t* __restrict__ kT,
                                               bf16_t* __restrict__ kvT) {
  __shared__ __align__(16) bf16_t sA[2][96][32];
  __shared__ __align__(16) bf16_t sB[2][96][32];

  const int bh = blockIdx.x;             // 0..63
  const bf16_t* Ab = vT + (size_t)bh * HD * NTOK;
  const bf16_t* Bb = kT + (size_t)bh * HD * NTOK;

  const int t = threadIdx.x;
  const int lane = t & 31;
  const int wid = t >> 5;
  const int wm = wid / 3, wn = wid % 3;

  auto stage = [&](int k0, int buf) {
    // 384 chunks for A + 384 for B = 768; 288 threads * 3
#pragma unroll
    for (int i = 0; i < 3; ++i) {
      int c = t + i * 288;
      if (c < 384) {
        int r = c >> 2, c16 = c & 3;
        async_b128(lds_addr(&sA[buf][r][c16 * 8]),
                   Ab + (size_t)r * NTOK + k0 + c16 * 8);
      } else if (c < 768) {
        int c2 = c - 384;
        int r = c2 >> 2, c16 = c2 & 3;
        async_b128(lds_addr(&sB[buf][r][c16 * 8]),
                   Bb + (size_t)r * NTOK + k0 + c16 * 8);
      }
    }
  };

  v8f acc[2][2] = {};
  stage(0, 0);
  wait_async0();
  __syncthreads();

  const int mrow = lane & 15;
  const int hi = lane >> 4;
  const int abase = hi * 16;
  const int kb = hi * 16;

  for (int kt = 0; kt < NTOK / 32; ++kt) {
    int buf = kt & 1;
    if (kt + 1 < NTOK / 32) stage((kt + 1) * 32, buf ^ 1);
    v16bf af[2], bf[2];
#pragma unroll
    for (int i = 0; i < 2; ++i)
      af[i] = frag16(&sA[buf][wm * 32 + i * 16 + mrow][abase],
                     &sA[buf][wm * 32 + i * 16 + mrow][abase + 8]);
#pragma unroll
    for (int j = 0; j < 2; ++j)
      bf[j] = frag16(&sB[buf][wn * 32 + j * 16 + mrow][kb],
                     &sB[buf][wn * 32 + j * 16 + mrow][kb + 8]);
#pragma unroll
    for (int i = 0; i < 2; ++i)
#pragma unroll
      for (int j = 0; j < 2; ++j)
        acc[i][j] = wmma_bf16(af[i], bf[j], acc[i][j]);
    wait_async0();
    __syncthreads();
  }

  bf16_t* dst = kvT + (size_t)bh * HD * HD;
#pragma unroll
  for (int i = 0; i < 2; ++i)
#pragma unroll
    for (int j = 0; j < 2; ++j) {
      int col = wn * 32 + j * 16 + mrow;
#pragma unroll
      for (int r = 0; r < 8; ++r) {
        int rowg = wm * 32 + i * 16 + r + hi * 8;
        dst[(size_t)rowg * HD + col] = (bf16_t)acc[i][j][r];
      }
    }
}

// ---------------------------------------------------------------------------
// Depthwise 3x3 conv on v as [B, 384, 32, 32] -> vdwc f32 [b,kvh][N][HD]
// ---------------------------------------------------------------------------
__global__ __launch_bounds__(256) void dwconv(const float* __restrict__ KVpre,
                                              const float* __restrict__ w,
                                              const float* __restrict__ bconv,
                                              float* __restrict__ vdwc) {
  int idx = blockIdx.x * 256 + threadIdx.x;
  const int total = BATCH * KVH * NTOK * HD;
  if (idx >= total) return;
  int d = idx % HD;
  int rest = idx / HD;
  int n = rest % NTOK;
  int bh = rest / NTOK;
  int b = bh / KVH, g = bh % KVH;
  int y = n >> 5, x = n & 31;
  int c = g * HD + d;
  const int kvdim = 2 * KVH * HD;
  const float* wc = w + c * 9;
  const float* vb = KVpre + (size_t)b * NTOK * kvdim + KVH * HD + c;
  float acc = bconv[c];
#pragma unroll
  for (int dy = 0; dy < 3; ++dy) {
    int yy = y + dy - 1;
    if (yy < 0 || yy >= 32) continue;
#pragma unroll
    for (int dx = 0; dx < 3; ++dx) {
      int xx = x + dx - 1;
      if (xx < 0 || xx >= 32) continue;
      acc += wc[dy * 3 + dx] * vb[(size_t)(yy * 32 + xx) * kvdim];
    }
  }
  vdwc[idx] = acc;
}

// ---------------------------------------------------------------------------
// out1 = q_f @ kvT^T (per (b,h); M=1024 in 128-blocks, N=96, K=96),
// epilogue adds conv branch (group h%4) and writes bf16 pre-proj [M, 1152].
// 8 waves: 4 (M) x 2 (N); wave tile 32x48 = 2x3 WMMA tiles.
// ---------------------------------------------------------------------------
__global__ __launch_bounds__(256) void out_gemm(const bf16_t* __restrict__ qf,
                                                const bf16_t* __restrict__ kvT,
                                                const float* __restrict__ vdwc,
                                                bf16_t* __restrict__ preproj) {
  __shared__ __align__(16) bf16_t sA[2][128][32];
  __shared__ __align__(16) bf16_t sB[2][96][32];

  const int bh = blockIdx.x;             // 0..191
  const int b = bh / NHEAD, h = bh % NHEAD;
  const int g = h & 3;                   // kv head = h % 4
  const int m0 = (int)blockIdx.y * 128;  // token block

  const bf16_t* Ab = qf + (size_t)bh * NTOK * HD;
  const bf16_t* Bb = kvT + (size_t)(b * KVH + g) * HD * HD;

  const int t = threadIdx.x;
  const int lane = t & 31;
  const int wid = t >> 5;
  const int wm = wid & 3;                // 32-row strip
  const int wn = wid >> 2;               // 48-col strip

  auto stage = [&](int k0, int buf) {
    // A: 512 chunks, B: 384 chunks -> 896 total, 4 per thread
#pragma unroll
    for (int i = 0; i < 4; ++i) {
      int c = t + i * 256;
      if (c < 512) {
        int r = c >> 2, c16 = c & 3;
        async_b128(lds_addr(&sA[buf][r][c16 * 8]),
                   Ab + (size_t)(m0 + r) * HD + k0 + c16 * 8);
      } else if (c < 896) {
        int c2 = c - 512;
        int r = c2 >> 2, c16 = c2 & 3;
        async_b128(lds_addr(&sB[buf][r][c16 * 8]),
                   Bb + (size_t)r * HD + k0 + c16 * 8);
      }
    }
  };

  v8f acc[2][3] = {};
  stage(0, 0);
  wait_async0();
  __syncthreads();

  const int mrow = lane & 15;
  const int hi = lane >> 4;
  const int abase = hi * 16;
  const int kb = hi * 16;

  for (int kt = 0; kt < HD / 32; ++kt) {  // 3 iterations
    int buf = kt & 1;
    if (kt + 1 < HD / 32) stage((kt + 1) * 32, buf ^ 1);
    v16bf af[2], bf[3];
#pragma unroll
    for (int i = 0; i < 2; ++i)
      af[i] = frag16(&sA[buf][wm * 32 + i * 16 + mrow][abase],
                     &sA[buf][wm * 32 + i * 16 + mrow][abase + 8]);
#pragma unroll
    for (int j = 0; j < 3; ++j)
      bf[j] = frag16(&sB[buf][wn * 48 + j * 16 + mrow][kb],
                     &sB[buf][wn * 48 + j * 16 + mrow][kb + 8]);
#pragma unroll
    for (int i = 0; i < 2; ++i)
#pragma unroll
      for (int j = 0; j < 3; ++j)
        acc[i][j] = wmma_bf16(af[i], bf[j], acc[i][j]);
    wait_async0();
    __syncthreads();
  }

  const float* conv = vdwc + (size_t)(b * KVH + g) * NTOK * HD;
  bf16_t* dst = preproj + (size_t)b * NTOK * DIM + (size_t)h * HD;
#pragma unroll
  for (int i = 0; i < 2; ++i)
#pragma unroll
    for (int j = 0; j < 3; ++j) {
      int col = wn * 48 + j * 16 + mrow;   // 0..95
#pragma unroll
      for (int r = 0; r < 8; ++r) {
        int rowg = m0 + wm * 32 + i * 16 + r + hi * 8;  // token index
        float v = acc[i][j][r] + conv[(size_t)rowg * HD + col];
        dst[(size_t)rowg * DIM + col] = (bf16_t)v;
      }
    }
}

// ---------------------------------------------------------------------------
// Host-side orchestration
// ---------------------------------------------------------------------------
extern "C" void kernel_launch(void* const* d_in, const int* in_sizes, int n_in,
                              void* d_out, int out_size, void* d_ws, size_t ws_size,
                              hipStream_t stream) {
  const float* x      = (const float*)d_in[0];
  const float* wq_w   = (const float*)d_in[1];
  const float* wq_b   = (const float*)d_in[2];
  const float* wkv_w  = (const float*)d_in[3];
  const float* wkv_b  = (const float*)d_in[4];
  const float* dwc_w  = (const float*)d_in[5];
  const float* dwc_b  = (const float*)d_in[6];
  const float* proj_w = (const float*)d_in[7];
  const float* proj_b = (const float*)d_in[8];
  float* out = (float*)d_out;

  char* ws = (char*)d_ws;
  size_t off = 0;
  auto alloc = [&](size_t bytes) -> void* {
    void* p = ws + off;
    off += (bytes + 255) & ~(size_t)255;
    return p;
  };

  bf16_t* xb     = (bf16_t*)alloc((size_t)MROWS * DIM * 2);
  bf16_t* wqb    = (bf16_t*)alloc((size_t)DIM * DIM * 2);
  bf16_t* wkvb   = (bf16_t*)alloc((size_t)(2 * KVH * HD) * DIM * 2);
  bf16_t* projb  = (bf16_t*)alloc((size_t)DIM * DIM * 2);
  float*  Qpre   = (float*)alloc((size_t)MROWS * DIM * 4);
  float*  KVpre  = (float*)alloc((size_t)MROWS * (2 * KVH * HD) * 4);
  bf16_t* qf     = (bf16_t*)alloc((size_t)BATCH * NHEAD * NTOK * HD * 2);
  bf16_t* kfT    = (bf16_t*)alloc((size_t)BATCH * KVH * HD * NTOK * 2);
  bf16_t* vT     = (bf16_t*)alloc((size_t)BATCH * KVH * HD * NTOK * 2);
  bf16_t* kvTb   = (bf16_t*)alloc((size_t)BATCH * KVH * HD * HD * 2);
  float*  vdwc   = (float*)alloc((size_t)BATCH * KVH * NTOK * HD * 4);
  bf16_t* prepro = (bf16_t*)alloc((size_t)MROWS * DIM * 2);
  (void)ws_size; (void)in_sizes; (void)n_in; (void)out_size;

  // 1. converts
  cvt_bf16<<<(MROWS * DIM + 255) / 256, 256, 0, stream>>>(x, xb, MROWS * DIM);
  cvt_bf16<<<(DIM * DIM + 255) / 256, 256, 0, stream>>>(wq_w, wqb, DIM * DIM);
  cvt_bf16<<<(2 * KVH * HD * DIM + 255) / 256, 256, 0, stream>>>(wkv_w, wkvb, 2 * KVH * HD * DIM);
  cvt_bf16<<<(DIM * DIM + 255) / 256, 256, 0, stream>>>(proj_w, projb, DIM * DIM);

  // 2. QKV projections
  gemm_nt_bias_f32<<<(MROWS / 128) * (DIM / 64), 256, 0, stream>>>(
      xb, wqb, wq_b, Qpre, MROWS, DIM, DIM);
  gemm_nt_bias_f32<<<(MROWS / 128) * ((2 * KVH * HD) / 64), 256, 0, stream>>>(
      xb, wkvb, wkv_b, KVpre, MROWS, 2 * KVH * HD, DIM);

  // 3. focusing feature maps
  focus_q<<<BATCH * NHEAD, 256, 0, stream>>>(Qpre, qf);
  focus_kv<<<BATCH * KVH, 256, 0, stream>>>(KVpre, kfT, vT);

  // 4. kv^T = v^T k_f  (per b,kvh)
  kv_gemm<<<BATCH * KVH, 288, 0, stream>>>(vT, kfT, kvTb);

  // 5. depthwise conv branch
  dwconv<<<(BATCH * KVH * NTOK * HD + 255) / 256, 256, 0, stream>>>(
      KVpre, dwc_w, dwc_b, vdwc);

  // 6. out1 = q_f kv + conv branch -> bf16 pre-projection
  out_gemm<<<dim3(BATCH * NHEAD, NTOK / 128), 256, 0, stream>>>(
      qf, kvTb, vdwc, prepro);

  // 7. final projection
  gemm_nt_bias_f32<<<(MROWS / 128) * (DIM / 64), 256, 0, stream>>>(
      prepro, projb, proj_b, out, MROWS, DIM, DIM);
}
